// FusionOp_47090021433860
// MI455X (gfx1250) — 8-bit matrix operand order repaired and checked by reading, compiled, not run on hardware
//
#include <hip/hip_runtime.h>
#include <stdint.h>


namespace {
constexpr int kE     = 32;
constexpr int kCap   = 48;
constexpr int kInter = 1408;
constexpr int kH     = 2048;
constexpr int kT     = 64;
constexpr int kTopK  = 8;
constexpr int kRows  = kT * kTopK;
constexpr int kSlots = kE * kCap;
constexpr int kN1    = 2 * kInter;

constexpr size_t OFF_SLOT  = 0;
constexpr size_t OFF_ROWOF = 4096;
constexpr size_t OFF_CNT   = OFF_ROWOF + 8192;
constexpr size_t OFF_S1    = OFF_CNT + 1024;
constexpr size_t OFF_S2    = OFF_S1 + (size_t)kSlots * 4;
constexpr size_t OFF_Q1    = ((OFF_S2 + (size_t)kSlots * 4 + 255) / 256) * 256;
constexpr size_t OFF_Q2    = OFF_Q1 + (size_t)kSlots * kH;
constexpr size_t OFF_H     = ((OFF_Q2 + (size_t)kSlots * kInter + 255) / 256) * 256;
constexpr size_t OFF_Y     = OFF_H + (size_t)kSlots * kN1 * 4;
}

typedef int v8i __attribute__((ext_vector_type(8)));

__global__ void route_kernel(const int* __restrict__ expert_ids,
                             int* __restrict__ slot,
                             int* __restrict__ row_of,
                             int* __restrict__ cnt) {
  int tid = threadIdx.x;
  for (int i = tid; i < kSlots; i += blockDim.x) row_of[i] = -1;
  __syncthreads();
  if (tid == 0) {
    int c[kE];
    for (int e = 0; e < kE; ++e) c[e] = 0;
    for (int r = 0; r < kRows; ++r) {
      int e = expert_ids[r];
      int p = c[e]++;
      if (p < kCap) {
        slot[r] = p;
        row_of[e * kCap + p] = r / kTopK;
      } else {
        slot[r] = -1;
      }
    }
    for (int e = 0; e < kE; ++e) cnt[e] = c[e] < kCap ? c[e] : kCap;
  }
}

__global__ __launch_bounds__(256)
void quant1_kernel(const float* __restrict__ x,
                   const float* __restrict__ smooth,
                   const int* __restrict__ row_of,
                   int8_t* __restrict__ q1,
                   float* __restrict__ s1) {
  int b = blockIdx.x;
  int e = b / kCap;
  int tid = threadIdx.x;
  __shared__ float red[256];
  __shared__ float s_sh;
  int t = row_of[b];
  float vals[8];
  float m = 0.f;
  if (t >= 0) {
    const float* xr = x + (size_t)t * kH;
    const float* sm = smooth + (size_t)e * kH;
#pragma unroll
    for (int j = 0; j < 8; ++j) {
      int h = tid + j * 256;
      float v = xr[h] * sm[h];
      vals[j] = v;
      m = fmaxf(m, fabsf(v));
    }
  } else {
#pragma unroll
    for (int j = 0; j < 8; ++j) vals[j] = 0.f;
  }
  red[tid] = m;
  __syncthreads();
  for (int s = 128; s > 0; s >>= 1) {
    if (tid < s) red[tid] = fmaxf(red[tid], red[tid + s]);
    __syncthreads();
  }
  if (tid == 0) {
    float sc = red[0] * (1.f / 127.f);
    if (sc == 0.f) sc = 1.f;
    s_sh = sc;
    s1[b] = sc;
  }
  __syncthreads();
  float inv = 1.f / s_sh;
  int8_t* qr = q1 + (size_t)b * kH;
#pragma unroll
  for (int j = 0; j < 8; ++j) {
    int h = tid + j * 256;
    float qv = rintf(vals[j] * inv);
    qv = fminf(fmaxf(qv, -128.f), 127.f);
    qr[h] = (int8_t)(int)qv;
  }
}

__device__ __forceinline__ v8i load_a_frag(const int8_t* __restrict__ arow,
                                           int k0, int g) {
  const int8_t* p = arow + k0 + g * 8;
  uint2 u0 = *(const uint2*)(p + 0);
  uint2 u1 = *(const uint2*)(p + 16);
  uint2 u2 = *(const uint2*)(p + 32);
  uint2 u3 = *(const uint2*)(p + 48);
  v8i a;
  a[0] = (int)u0.x; a[1] = (int)u0.y;
  a[2] = (int)u1.x; a[3] = (int)u1.y;
  a[4] = (int)u2.x; a[5] = (int)u2.y;
  a[6] = (int)u3.x; a[7] = (int)u3.y;
  return a;
}

__device__ __forceinline__ void load_braw(const int* __restrict__ wcol, int k0,
                                          int g, int Ndim, int* __restrict__ w) {
#pragma unroll
  for (int v = 0; v < 8; ++v) {
    int kb = k0 + g * 8 + (v >> 1) * 16 + (v & 1) * 4;
    const int* p = wcol + (size_t)kb * Ndim;
    w[v * 4 + 0] = p[0];
    w[v * 4 + 1] = p[Ndim];
    w[v * 4 + 2] = p[2 * Ndim];
    w[v * 4 + 3] = p[3 * Ndim];
  }
}

__device__ __forceinline__ v8i pack_b(const int* __restrict__ w) {
  v8i B;
#pragma unroll
  for (int v = 0; v < 8; ++v) {
    uint32_t pk = (uint32_t)((w[4 * v + 0] - 63) & 0xFF) |
                  ((uint32_t)((w[4 * v + 1] - 63) & 0xFF) << 8) |
                  ((uint32_t)((w[4 * v + 2] - 63) & 0xFF) << 16) |
                  ((uint32_t)((w[4 * v + 3] - 63) & 0xFF) << 24);
    B[v] = (int)pk;
  }
  return B;
}

template <int NM>
__device__ __forceinline__ void moe_gemm_body(
    const int8_t* __restrict__ qact, const int* __restrict__ wq,
    const float* __restrict__ srow, const float* __restrict__ scol,
    float* __restrict__ outbuf, int Kdim, int Ndim, int e) {
  int lane = threadIdx.x & 31;
  int wave = threadIdx.x >> 5;
  int l16 = lane & 15;
  int g   = lane >> 4;
  int col = blockIdx.x * 64 + wave * 16 + l16;

  const int* __restrict__ wcol = wq + (size_t)e * Kdim * Ndim + col;
  const int8_t* arow[3];
  arow[0] = qact + (size_t)e * kCap * Kdim + (size_t)l16 * Kdim;
  arow[1] = arow[0] + (size_t)16 * Kdim;
  arow[2] = arow[0] + (size_t)32 * Kdim;

  v8i acc[NM];
#pragma unroll
  for (int m = 0; m < NM; ++m) acc[m] = (v8i){};

  int wbuf0[32], wbuf1[32];
  load_braw(wcol, 0, g, Ndim, wbuf0);

  for (int k0 = 0; k0 < Kdim; k0 += 128) {
    if (k0 + 256 < Kdim)
      __builtin_prefetch(wcol + (size_t)(k0 + 256) * Ndim, 0, 3);

    load_braw(wcol, k0 + 64, g, Ndim, wbuf1);
    {
      v8i B = pack_b(wbuf0);
#pragma unroll
      for (int m = 0; m < NM; ++m) {
        v8i A = load_a_frag(arow[m], k0, g);
        acc[m] = __builtin_amdgcn_wmma_i32_16x16x64_iu8(true, A, true, B,
                                                        acc[m], false, false);
      }
    }
    if (k0 + 128 < Kdim)
      load_braw(wcol, k0 + 128, g, Ndim, wbuf0);
    {
      v8i B = pack_b(wbuf1);
#pragma unroll
      for (int m = 0; m < NM; ++m) {
        v8i A = load_a_frag(arow[m], k0 + 64, g);
        acc[m] = __builtin_amdgcn_wmma_i32_16x16x64_iu8(true, A, true, B,
                                                        acc[m], false, false);
      }
    }
  }

  float sc_col = scol[(size_t)e * Ndim + col];
#pragma unroll
  for (int m = 0; m < NM; ++m) {
#pragma unroll
    for (int v = 0; v < 8; ++v) {
      int row = m * 16 + v + g * 8;
      float val = (float)acc[m][v] * srow[e * kCap + row] * sc_col;
      outbuf[(size_t)(e * kCap + row) * Ndim + col] = val;
    }
  }
}

__global__ __launch_bounds__(128)
void moe_gemm_i8_kernel(const int8_t* __restrict__ qact,
                        const int* __restrict__ wq,
                        const float* __restrict__ srow,
                        const float* __restrict__ scol,
                        const int* __restrict__ cnt,
                        float* __restrict__ outbuf,
                        int Kdim, int Ndim) {
  int e = blockIdx.z;
  int c = __builtin_amdgcn_readfirstlane(cnt[e]);
  if (c <= 0) return;
  int nm = (c + 15) >> 4;
  if (nm == 1)      moe_gemm_body<1>(qact, wq, srow, scol, outbuf, Kdim, Ndim, e);
  else if (nm == 2) moe_gemm_body<2>(qact, wq, srow, scol, outbuf, Kdim, Ndim, e);
  else              moe_gemm_body<3>(qact, wq, srow, scol, outbuf, Kdim, Ndim, e);
}

__global__ __launch_bounds__(256)
void act_quant_kernel(const float* __restrict__ hbuf,
                      const int* __restrict__ cnt,
                      int8_t* __restrict__ q2,
                      float* __restrict__ s2) {
  int b = blockIdx.x;
  int e = b / kCap;
  int slot = b % kCap;
  int c = cnt[e];
  int cpad = (c + 15) & ~15;
  if (cpad > kCap) cpad = kCap;
  if (slot >= cpad) return;
  int tid = threadIdx.x;
  __shared__ float red[256];
  __shared__ float s_sh;
  const float* hr = hbuf + (size_t)b * kN1;
  float av[6];
  float m = 0.f;
#pragma unroll
  for (int j = 0; j < 6; ++j) {
    int h = tid + j * 256;
    float aa = 0.f;
    if (h < kInter) {
      float gt = hr[h];
      float up = hr[h + kInter];
      aa = (gt / (1.f + expf(-gt))) * up;
      m = fmaxf(m, fabsf(aa));
    }
    av[j] = aa;
  }
  red[tid] = m;
  __syncthreads();
  for (int s = 128; s > 0; s >>= 1) {
    if (tid < s) red[tid] = fmaxf(red[tid], red[tid + s]);
    __syncthreads();
  }
  if (tid == 0) {
    float sc = red[0] * (1.f / 127.f);
    if (sc == 0.f) sc = 1.f;
    s_sh = sc;
    s2[b] = sc;
  }
  __syncthreads();
  float inv = 1.f / s_sh;
  int8_t* qr = q2 + (size_t)b * kInter;
#pragma unroll
  for (int j = 0; j < 6; ++j) {
    int h = tid + j * 256;
    if (h < kInter) {
      float qv = rintf(av[j] * inv);
      qv = fminf(fmaxf(qv, -128.f), 127.f);
      qr[h] = (int8_t)(int)qv;
    }
  }
}

__global__ __launch_bounds__(256)
void gather_kernel(const int* __restrict__ expert_ids,
                   const float* __restrict__ expert_scales,
                   const int* __restrict__ slot,
                   const float* __restrict__ ybuf,
                   float* __restrict__ out) {
  int t = blockIdx.x;
  int tid = threadIdx.x;
#pragma unroll
  for (int j = 0; j < 8; ++j) {
    int h = tid + j * 256;
    float acc = 0.f;
    for (int k = 0; k < kTopK; ++k) {
      int r = t * kTopK + k;
      int s = slot[r];
      if (s >= 0) {
        int e = expert_ids[r];
        acc += expert_scales[r] * ybuf[(size_t)(e * kCap + s) * kH + h];
      }
    }
    out[(size_t)t * kH + h] = acc;
  }
}

extern "C" void kernel_launch(void* const* d_in, const int* in_sizes, int n_in,
                              void* d_out, int out_size, void* d_ws, size_t ws_size,
                              hipStream_t stream) {
  const float* x       = (const float*)d_in[0];
  const int*   eids    = (const int*)d_in[1];
  const int*   w1      = (const int*)d_in[2];
  const float* w1scale = (const float*)d_in[3];
  const int*   w2      = (const int*)d_in[4];
  const float* w2scale = (const float*)d_in[5];
  const float* smooth  = (const float*)d_in[6];
  const float* escale  = (const float*)d_in[7];
  float* out = (float*)d_out;

  char* wsb = (char*)d_ws;
  int*     slot   = (int*)(wsb + OFF_SLOT);
  int*     row_of = (int*)(wsb + OFF_ROWOF);
  int*     cnt    = (int*)(wsb + OFF_CNT);
  float*   s1     = (float*)(wsb + OFF_S1);
  float*   s2     = (float*)(wsb + OFF_S2);
  int8_t*  q1     = (int8_t*)(wsb + OFF_Q1);
  int8_t*  q2     = (int8_t*)(wsb + OFF_Q2);
  float*   hbuf   = (float*)(wsb + OFF_H);
  float*   ybuf   = (float*)(wsb + OFF_Y);

  route_kernel<<<1, 512, 0, stream>>>(eids, slot, row_of, cnt);
  quant1_kernel<<<kSlots, 256, 0, stream>>>(x, smooth, row_of, q1, s1);
  moe_gemm_i8_kernel<<<dim3(kN1 / 64, 1, kE), 128, 0, stream>>>(
      q1, w1, s1, w1scale, cnt, hbuf, kH, kN1);
  act_quant_kernel<<<kSlots, 256, 0, stream>>>(hbuf, cnt, q2, s2);
  moe_gemm_i8_kernel<<<dim3(kH / 64, 1, kE), 128, 0, stream>>>(
      q2, w2, s2, w2scale, cnt, ybuf, kInter, kH);
  gather_kernel<<<kT, 256, 0, stream>>>(eids, escale, slot, ybuf, out);
}
